// TestSiluMulGroupFp8QuantModel_1374389534753
// MI455X (gfx1250) — 8-bit matrix operand order repaired and checked by reading, compiled, not run on hardware
//
#include <hip/hip_runtime.h>
#include <hip/hip_bf16.h>
#include <stdint.h>

#define M_DIM 8192
#define N_DIM 4096
#define K_DIM 4096
#define GROUP 128
#define NKG   (K_DIM / GROUP)
#define NNG   (N_DIM / GROUP)
#define FP8_MAX 448.0f

typedef __attribute__((ext_vector_type(16))) int   v16i;
typedef __attribute__((ext_vector_type(8)))  float v8f;
typedef __attribute__((ext_vector_type(4)))  int   v4i_t;

#define AS1 __attribute__((address_space(1)))
#define AS3 __attribute__((address_space(3)))

#if defined(__has_builtin)
#  if __has_builtin(__builtin_amdgcn_global_load_async_to_lds_b128)
#    define HAVE_ASYNC_LDS 1
#  endif
#  if __has_builtin(__builtin_amdgcn_s_wait_asynccnt)
#    define HAVE_WAIT_ASYNC 1
#  endif
#  if __has_builtin(__builtin_amdgcn_cvt_pk_fp8_f32)
#    define HAVE_CVT_PK_FP8 1
#  endif
#endif
#ifndef HAVE_ASYNC_LDS
#  define HAVE_ASYNC_LDS 0
#endif
#ifndef HAVE_WAIT_ASYNC
#  define HAVE_WAIT_ASYNC 0
#endif
#ifndef HAVE_CVT_PK_FP8
#  define HAVE_CVT_PK_FP8 0
#endif

__device__ __forceinline__ unsigned f32_to_e4m3(float f) {
  unsigned u = __float_as_uint(f);
  unsigned s = (u >> 24) & 0x80u;
  unsigned a = u & 0x7fffffffu;
  if (a > 0x43e00000u) a = 0x43e00000u;
  unsigned rnd = a + 0x0007ffffu + ((a >> 20) & 1u);
  int e = (int)(rnd >> 23) - 127;
  unsigned m = (rnd >> 20) & 7u;
  unsigned r;
  if (e < -9) {
    r = 0u;
  } else if (e < -6) {
    int sh = -6 - e;
    r = (0x8u | m) >> sh;
  } else {
    r = ((unsigned)(e + 7) << 3) | m;
  }
  return s | r;
}

__device__ __forceinline__ unsigned pack_fp8x4(float a, float b, float c, float d) {
#if HAVE_CVT_PK_FP8
  int lo = __builtin_amdgcn_cvt_pk_fp8_f32(a, b, 0, false);
  int both = __builtin_amdgcn_cvt_pk_fp8_f32(c, d, lo, true);
  return (unsigned)both;
#else
  return f32_to_e4m3(a) | (f32_to_e4m3(b) << 8) |
         (f32_to_e4m3(c) << 16) | (f32_to_e4m3(d) << 24);
#endif
}

__device__ __forceinline__ void async_cp16(unsigned char* lds_dst,
                                           const unsigned char* gsrc) {
#if HAVE_ASYNC_LDS
  __builtin_amdgcn_global_load_async_to_lds_b128(
      (AS1 v4i_t*)gsrc, (AS3 v4i_t*)lds_dst, 0, 0);
#else
  *(uint4*)lds_dst = *(const uint4*)gsrc;
#endif
}

__device__ __forceinline__ void wait_async0() {
#if HAVE_WAIT_ASYNC
  __builtin_amdgcn_s_wait_asynccnt(0);
#elif HAVE_ASYNC_LDS
  asm volatile("s_wait_asynccnt 0" ::: "memory");
#endif
}

__global__ __launch_bounds__(256) void silu_mul_quant_kernel(
    const float* __restrict__ x,
    unsigned char* __restrict__ yq,
    float* __restrict__ ascale) {
  const int tid  = threadIdx.x;
  const int lane = tid & 31;
  const int wid  = tid >> 5;
  const int g    = blockIdx.x * 8 + wid;
  const int m    = g >> 5;
  const int kg   = g & 31;

  const float* gp = x + (size_t)m * (2 * K_DIM) + kg * GROUP + lane * 4;
  const float* up = gp + K_DIM;
  float4 gv = *(const float4*)gp;
  float4 uv = *(const float4*)up;

  float v0 = (gv.x / (1.0f + __expf(-gv.x))) * uv.x;
  float v1 = (gv.y / (1.0f + __expf(-gv.y))) * uv.y;
  float v2 = (gv.z / (1.0f + __expf(-gv.z))) * uv.z;
  float v3 = (gv.w / (1.0f + __expf(-gv.w))) * uv.w;

  float amax = fmaxf(fmaxf(fabsf(v0), fabsf(v1)), fmaxf(fabsf(v2), fabsf(v3)));
#pragma unroll
  for (int off = 16; off > 0; off >>= 1)
    amax = fmaxf(amax, __shfl_xor(amax, off, 32));

  float scale = fmaxf(amax, 1e-12f) * (1.0f / FP8_MAX);
  float inv   = 1.0f / scale;

  unsigned packed = pack_fp8x4(v0 * inv, v1 * inv, v2 * inv, v3 * inv);
  *(unsigned*)(yq + (size_t)m * K_DIM + kg * GROUP + lane * 4) = packed;
  if (lane == 0) ascale[(size_t)kg * M_DIM + m] = scale;
}

__global__ __launch_bounds__(256) void wq_convert_transpose_kernel(
    const float* __restrict__ wq,
    unsigned char* __restrict__ wt) {
  __shared__ unsigned char sm[32][36];
  const int tid = threadIdx.x;
  const int tx = tid & 31, ty = tid >> 5;
  const int n0 = blockIdx.x * 32, k0 = blockIdx.y * 32;
#pragma unroll
  for (int i = 0; i < 4; ++i) {
    int k = k0 + ty + i * 8;
    float val = wq[(size_t)k * N_DIM + n0 + tx];
    sm[tx][ty + i * 8] = (unsigned char)f32_to_e4m3(val);
  }
  __syncthreads();
  const int r = tid >> 3, c = tid & 7;
  unsigned p = (unsigned)sm[r][c * 4 + 0] |
               ((unsigned)sm[r][c * 4 + 1] << 8) |
               ((unsigned)sm[r][c * 4 + 2] << 16) |
               ((unsigned)sm[r][c * 4 + 3] << 24);
  *(unsigned*)(wt + (size_t)(n0 + r) * K_DIM + k0 + c * 4) = p;
}

__global__ __launch_bounds__(256) void fp8_gemm_kernel(
    const unsigned char* __restrict__ Aq,
    const float* __restrict__ Ascale,
    const unsigned char* __restrict__ Bt,
    const float* __restrict__ Wscale,
    float* __restrict__ Out) {
  __shared__ unsigned char smA[2][128 * 128];
  __shared__ unsigned char smB[2][128 * 128];

  const int nblk = blockIdx.x;
  const int mblk = blockIdx.y;
  const int m0 = mblk * 128, n0 = nblk * 128;
  const int tid  = threadIdx.x;
  const int lane = tid & 31;
  const int wid  = tid >> 5;
  const int l16  = lane & 15;
  const int hi   = (lane >> 4) & 1;
  const int wm   = (wid >> 1) * 32;
  const int wn   = (wid & 1) * 64;

  v8f acc[2][4] = {};

  auto issue = [&](int buf, int kg) {
#pragma unroll
    for (int i = 0; i < 4; ++i) {
      int o = tid * 16 + i * 4096;
      int row = o >> 7, col = o & 127;
      async_cp16(&smA[buf][o],
                 Aq + (size_t)(m0 + row) * K_DIM + kg * GROUP + col);
      async_cp16(&smB[buf][o],
                 Bt + (size_t)(n0 + row) * K_DIM + kg * GROUP + col);
    }
  };

  issue(0, 0);

  for (int kg = 0; kg < NKG; ++kg) {
    const int buf = kg & 1;
    wait_async0();
    __syncthreads();
    if (kg + 1 < NKG) issue(buf ^ 1, kg + 1);

    const float wsc = Wscale[kg * NNG + nblk];

    v8f t[2];
#pragma unroll
    for (int mt = 0; mt < 2; ++mt) {
      const float* sp = Ascale + (size_t)kg * M_DIM + m0 + wm + mt * 16 + hi * 8;
      float4 s_lo = *(const float4*)sp;
      float4 s_hi = *(const float4*)(sp + 4);
      t[mt][0] = s_lo.x * wsc; t[mt][1] = s_lo.y * wsc;
      t[mt][2] = s_lo.z * wsc; t[mt][3] = s_lo.w * wsc;
      t[mt][4] = s_hi.x * wsc; t[mt][5] = s_hi.y * wsc;
      t[mt][6] = s_hi.z * wsc; t[mt][7] = s_hi.w * wsc;
    }

    v16i af[2];
#pragma unroll
    for (int mt = 0; mt < 2; ++mt) {
      const unsigned char* p = &smA[buf][(wm + mt * 16 + l16) * 128 + hi * 16];
      union { v16i v; unsigned long long d[8]; } ua;
#pragma unroll
      for (int j = 0; j < 8; ++j)
        ua.d[j] = *(const unsigned long long*)(p + 32 * (j >> 1) + 8 * (j & 1));
      af[mt] = ua.v;
    }

#pragma unroll
    for (int nt = 0; nt < 4; ++nt) {
      const unsigned char* p = &smB[buf][(wn + nt * 16 + l16) * 128 + hi * 16];
      union { v16i v; uint4 q[4]; } ub;
#pragma unroll
      for (int c = 0; c < 4; ++c)
        ub.q[c] = *(const uint4*)(p + 32 * c);

#pragma unroll
      for (int mt = 0; mt < 2; ++mt) {
        v8f z = {};
        v8f pr = __builtin_amdgcn_wmma_f32_16x16x128_fp8_fp8(
            af[mt], ub.v, (short)0, z, false, false);
#pragma unroll
        for (int r = 0; r < 8; ++r)
          acc[mt][nt][r] += t[mt][r] * pr[r];
      }
    }
  }

#pragma unroll
  for (int mt = 0; mt < 2; ++mt)
#pragma unroll
    for (int nt = 0; nt < 4; ++nt)
#pragma unroll
      for (int r = 0; r < 8; ++r) {
        int m = m0 + wm + mt * 16 + hi * 8 + r;
        int n = n0 + wn + nt * 16 + l16;
        Out[(size_t)m * N_DIM + n] = acc[mt][nt][r];
      }
}

extern "C" void kernel_launch(void* const* d_in, const int* in_sizes, int n_in,
                              void* d_out, int out_size, void* d_ws,
                              size_t ws_size, hipStream_t stream) {
  (void)in_sizes; (void)n_in; (void)out_size; (void)ws_size;
  const float* x      = (const float*)d_in[0];
  const float* w_q    = (const float*)d_in[1];
  const float* wscale = (const float*)d_in[2];
  float* out = (float*)d_out;

  unsigned char* ws = (unsigned char*)d_ws;
  unsigned char* yq = ws;
  float* ascale = (float*)(ws + (size_t)M_DIM * K_DIM);
  unsigned char* wt = ws + (size_t)M_DIM * K_DIM
                         + (size_t)NKG * M_DIM * sizeof(float);

  silu_mul_quant_kernel<<<(M_DIM * NKG) / 8, 256, 0, stream>>>(x, yq, ascale);
  wq_convert_transpose_kernel<<<dim3(N_DIM / 32, K_DIM / 32), 256, 0, stream>>>(
      w_q, wt);
  fp8_gemm_kernel<<<dim3(N_DIM / 128, M_DIM / 128), 256, 0, stream>>>(
      yq, ascale, wt, wscale, out);
}
